// Tagger_14370960572598
// MI455X (gfx1250) — compile-verified, matrix operand order checked by reading, not run on hardware
//
#include <hip/hip_runtime.h>

// ---------------- problem constants ----------------
static constexpr int L = 4096;   // sequence length
static constexpr int E = 512;    // embed dim
static constexpr int H = 512;    // hidden
static constexpr int G = 4 * H;  // 2048 gate rows
static constexpr int T = 32;     // tags
static constexpr int TAG_END = 1;

// ---------------- WMMA vector types ----------------
typedef __attribute__((ext_vector_type(16))) __bf16        v16bf;
typedef __attribute__((ext_vector_type(8)))  float         v8f;
typedef __attribute__((ext_vector_type(4)))  unsigned int  uv4;
typedef __attribute__((ext_vector_type(2)))  unsigned int  uv2;

union ABu { uv4 u[2]; v16bf v; };

// float -> bf16 (round to nearest even), pure bit ops
__device__ __forceinline__ unsigned short f2bf(float f) {
    unsigned u = __float_as_uint(f);
    u += 0x7FFFu + ((u >> 16) & 1u);
    return (unsigned short)(u >> 16);
}

__device__ __forceinline__ float sigmoidf_(float x) {
    return 1.0f / (1.0f + __expf(-x));
}

// ---------------- prep kernels ----------------
__global__ __launch_bounds__(256) void k_embed_gather(
    const int* __restrict__ sent, const float* __restrict__ embed,
    unsigned short* __restrict__ xs_bf)
{
    int idx = blockIdx.x * blockDim.x + threadIdx.x;
    if (idx >= L * E) return;
    int l = idx >> 9;          // / E
    int e = idx & (E - 1);
    xs_bf[idx] = f2bf(embed[(size_t)sent[l] * E + e]);
}

__global__ __launch_bounds__(256) void k_cast_bf16(
    const float* __restrict__ src, unsigned short* __restrict__ dst, int n)
{
    int idx = blockIdx.x * blockDim.x + threadIdx.x;
    if (idx < n) dst[idx] = f2bf(src[idx]);
}

__global__ __launch_bounds__(256) void k_bias_sum(
    const float* __restrict__ bihf, const float* __restrict__ bhhf,
    const float* __restrict__ bihb, const float* __restrict__ bhhb,
    float* __restrict__ bias)   // (2, 2048)
{
    int i = blockIdx.x * blockDim.x + threadIdx.x;
    if (i < G)            bias[i] = bihf[i] + bhhf[i];
    else if (i < 2 * G)   bias[i] = bihb[i - G] + bhhb[i - G];
}

// ---------------- WMMA GEMM:  C(M,N) = A(M,K) @ B(N,K)^T + bias(N) ----------------
// Each wave computes a 16 x (16*NT) strip: the A fragment is loaded once per
// k-chunk and reused for NT v_wmma_f32_16x16x32_bf16 issues (A-reuse cuts
// global traffic from 4*NT to 2+2*NT loads per NT WMMAs).
template <int NT>
__global__ __launch_bounds__(256) void k_wmma_gemm(
    const unsigned short* __restrict__ A,   // (M,K) bf16 row-major
    const unsigned short* __restrict__ B,   // (N,K) bf16 row-major
    const float* __restrict__ bias,         // (N)
    float* __restrict__ C,                  // (M,N)
    int K, int N, int groupsN)              // groupsN = (N/16)/NT
{
    const int lane = threadIdx.x & 31;
    const int wave = threadIdx.x >> 5;
    const int grp  = blockIdx.x * (blockDim.x >> 5) + wave;
    const int tm   = grp / groupsN;
    const int tn0  = (grp % groupsN) * NT;

    // 16-bit operand lane layout (ISA 7.12.2):
    // lanes 0-15: row = lane,     K chunks {0..7, 16..23}
    // lanes 16-31: row = lane-16, K chunks {8..15, 24..31}
    const int rsel = lane & 15;
    const int base = (lane < 16) ? 0 : 8;
    const size_t arow = (size_t)(tm * 16 + rsel) * K;

    v8f acc[NT];
    #pragma unroll
    for (int nt = 0; nt < NT; ++nt) acc[nt] = (v8f){};

    for (int k0 = 0; k0 < K; k0 += 32) {
        ABu a;
        const unsigned short* ap = A + arow + k0 + base;
        a.u[0] = *(const uv4*)(ap);        // K = k0+base    .. +7
        a.u[1] = *(const uv4*)(ap + 16);   // K = k0+16+base .. +7
        #pragma unroll
        for (int nt = 0; nt < NT; ++nt) {
            ABu b;
            const unsigned short* bp =
                B + (size_t)((tn0 + nt) * 16 + rsel) * K + k0 + base;
            b.u[0] = *(const uv4*)(bp);
            b.u[1] = *(const uv4*)(bp + 16);
            acc[nt] = __builtin_amdgcn_wmma_f32_16x16x32_bf16(
                false, a.v, false, b.v, (short)0, acc[nt], false, false);
        }
    }

    // C/D layout: element v -> M = tm*16 + v + (lane>=16 ? 8 : 0), N = tn*16 + (lane&15)
    const int mBase = tm * 16 + ((lane >> 4) << 3);
    #pragma unroll
    for (int nt = 0; nt < NT; ++nt) {
        const int n = (tn0 + nt) * 16 + rsel;
        const float bv = bias[n];
        #pragma unroll
        for (int v = 0; v < 8; ++v)
            C[(size_t)(mBase + v) * N + n] = acc[nt][v] + bv;
    }
}

// ---------------- LSTM recurrence ----------------
// 32 blocks: dir = blk>>4 (0=fwd,1=bwd), wg = blk&15 owns h indices [wg*32, wg*32+32)
// and the 4 matching gate-row blocks (i/f/g/o) => 128 rows x 512 K, bf16 in LDS (128 KB).
// LDS layout packs 4 consecutive K per row into one 8-byte entry so the inner
// loop is one conflict-free ds_load_b64 per 4 MACs.
// Cross-WG h exchange via double-buffered global hbuf + monotonic atomic barrier.
__global__ __launch_bounds__(128) void k_lstm(
    const float* __restrict__ xproj,      // (2, L, G) fp32, bias already folded
    const float* __restrict__ whh_f,
    const float* __restrict__ whh_b,
    const float* __restrict__ h0,         // (2, H)
    const float* __restrict__ c0,         // (2, H)
    float* __restrict__ hbuf,             // (2 dirs, 2 slots, H)
    unsigned short* __restrict__ hs,      // (L, 2H) bf16, cols [0,512)=fwd [512,1024)=bwd
    unsigned* __restrict__ cnt)           // (2) barrier counters
{
    extern __shared__ char smem[];
    unsigned short* wl  = (unsigned short*)smem;          // 128*512 bf16 (packed quads)
    const uv2*      wl2 = (const uv2*)smem;
    float* h_lds = (float*)(smem + 128 * 512 * 2);        // 512
    float* g_lds = h_lds + H;                             // 128

    const int tid = threadIdx.x;
    const int dir = blockIdx.x >> 4;
    const int wg  = blockIdx.x & 15;
    const int j0  = wg * 32;
    const float* whh = dir ? whh_b : whh_f;
    const float* xp  = xproj + (size_t)dir * L * G;

    // local row r = q*32 + jj  <->  global gate row q*512 + j0 + jj
    // packed layout: K quad (k..k+3) of row r at ushort index ((k>>2)*128 + r)*4 + (k&3)
    for (int i = tid; i < 128 * 512; i += 128) {
        int r = i & 127;
        int k = i >> 7;
        int q = r >> 5, jj = r & 31;
        float w = whh[(size_t)(q * 512 + j0 + jj) * H + k];
        wl[(((k >> 2) * 128 + r) << 2) + (k & 3)] = f2bf(w);
    }
    for (int k = tid; k < H; k += 128) h_lds[k] = h0[dir * H + k];
    float c_reg = 0.0f;
    if (tid < 32) c_reg = c0[dir * H + j0 + tid];
    __syncthreads();

    const int q  = tid >> 5;
    const int jj = tid & 31;
    const int grow = q * 512 + j0 + jj;

    for (int s = 0; s < L; ++s) {
        const int l = dir ? (L - 1 - s) : s;

        // gate = xproj[l][grow] + <W_hh row, h>
        float acc = xp[(size_t)l * G + grow];
        #pragma unroll 4
        for (int k = 0; k < H; k += 4) {
            uv2 u = wl2[(k >> 2) * 128 + tid];
            acc = fmaf(__uint_as_float(u.x << 16),         h_lds[k],     acc);
            acc = fmaf(__uint_as_float(u.x & 0xffff0000u), h_lds[k + 1], acc);
            acc = fmaf(__uint_as_float(u.y << 16),         h_lds[k + 2], acc);
            acc = fmaf(__uint_as_float(u.y & 0xffff0000u), h_lds[k + 3], acc);
        }
        g_lds[tid] = acc;
        __syncthreads();

        const int slot = (s + 1) & 1;
        if (tid < 32) {
            float iv = sigmoidf_(g_lds[tid]);
            float fv = sigmoidf_(g_lds[32 + tid]);
            float gv = tanhf(g_lds[64 + tid]);
            float ov = sigmoidf_(g_lds[96 + tid]);
            c_reg = fv * c_reg + iv * gv;
            float hv = ov * tanhf(c_reg);
            __hip_atomic_store(&hbuf[(dir * 2 + slot) * H + j0 + tid], hv,
                               __ATOMIC_RELAXED, __HIP_MEMORY_SCOPE_AGENT);
            hs[(size_t)l * (2 * H) + dir * H + j0 + tid] = f2bf(hv);
        }
        __threadfence();
        __syncthreads();
        if (tid == 0) {
            __hip_atomic_fetch_add(&cnt[dir], 1u, __ATOMIC_RELEASE,
                                   __HIP_MEMORY_SCOPE_AGENT);
            const unsigned target = 16u * (unsigned)(s + 1);
            while (__hip_atomic_load(&cnt[dir], __ATOMIC_ACQUIRE,
                                     __HIP_MEMORY_SCOPE_AGENT) < target) {
                __builtin_amdgcn_s_sleep(1);
            }
        }
        __syncthreads();
        for (int k = tid; k < H; k += 128)
            h_lds[k] = __hip_atomic_load(&hbuf[(dir * 2 + slot) * H + k],
                                         __ATOMIC_RELAXED, __HIP_MEMORY_SCOPE_AGENT);
        __syncthreads();
    }
}

// ---------------- Viterbi + backtrace (single wave) ----------------
__global__ __launch_bounds__(32) void k_viterbi(
    const float* __restrict__ feats,   // (L, T)
    const float* __restrict__ trans,   // (T, T) trans[next, prev]
    unsigned char* __restrict__ bptr,  // (L, T)
    float* __restrict__ out)           // [score, path(L)]
{
    __shared__ float fv_sh[T];
    const int t = threadIdx.x;

    float tr[T];
    #pragma unroll
    for (int p = 0; p < T; ++p) tr[p] = trans[t * T + p];

    fv_sh[t] = (t == 0) ? 0.0f : -10000.0f;   // START = 0
    __syncthreads();

    for (int l = 0; l < L; ++l) {
        float best = -3.4e38f; int arg = 0;
        #pragma unroll
        for (int p = 0; p < T; ++p) {
            float s = tr[p] + fv_sh[p];
            if (s > best) { best = s; arg = p; }   // first-max tie break
        }
        float fnew = best + feats[(size_t)l * T + t];
        bptr[(size_t)l * T + t] = (unsigned char)arg;
        __syncthreads();
        fv_sh[t] = fnew;
        __syncthreads();
    }

    if (t == 0) {
        float best = -3.4e38f; int last = 0;
        for (int p = 0; p < T; ++p) {
            float s = fv_sh[p] + trans[TAG_END * T + p];
            if (s > best) { best = s; last = p; }
        }
        out[0] = best;
        int tag = last;
        for (int i = L - 1; i >= 0; --i) {
            out[1 + i] = (float)tag;
            tag = (int)bptr[(size_t)i * T + tag];
        }
    }
}

// ---------------- host launcher ----------------
extern "C" void kernel_launch(void* const* d_in, const int* in_sizes, int n_in,
                              void* d_out, int out_size, void* d_ws, size_t ws_size,
                              hipStream_t stream)
{
    const int*   sent   = (const int*)  d_in[0];
    const float* embed  = (const float*)d_in[1];
    const float* w_ih_f = (const float*)d_in[2];
    const float* w_hh_f = (const float*)d_in[3];
    const float* b_ih_f = (const float*)d_in[4];
    const float* b_hh_f = (const float*)d_in[5];
    const float* w_ih_b = (const float*)d_in[6];
    const float* w_hh_b = (const float*)d_in[7];
    const float* b_ih_b = (const float*)d_in[8];
    const float* b_hh_b = (const float*)d_in[9];
    const float* w_out  = (const float*)d_in[10];
    const float* b_out  = (const float*)d_in[11];
    const float* h0     = (const float*)d_in[12];
    const float* c0     = (const float*)d_in[13];
    const float* trans  = (const float*)d_in[14];

    // workspace carve-up (256B aligned)
    char* ws = (char*)d_ws;
    size_t cur = 0;
    auto take = [&](size_t bytes) -> void* {
        void* p = ws + cur;
        cur += (bytes + 255) & ~(size_t)255;
        return p;
    };
    unsigned short* xs_bf   = (unsigned short*)take((size_t)L * E * 2);       // 4 MB
    unsigned short* wihf_bf = (unsigned short*)take((size_t)G * E * 2);       // 2 MB
    unsigned short* wihb_bf = (unsigned short*)take((size_t)G * E * 2);       // 2 MB
    unsigned short* wout_bf = (unsigned short*)take((size_t)T * 2 * H * 2);   // 64 KB
    float*          bias    = (float*)take((size_t)2 * G * 4);                // 16 KB
    float*          xproj   = (float*)take((size_t)2 * L * G * 4);            // 64 MB
    unsigned short* hs_bf   = (unsigned short*)take((size_t)L * 2 * H * 2);   // 8 MB
    float*          hbuf    = (float*)take((size_t)2 * 2 * H * 4);            // 8 KB
    float*          feats   = (float*)take((size_t)L * T * 4);                // 512 KB
    unsigned char*  bptr    = (unsigned char*)take((size_t)L * T);            // 128 KB
    unsigned*       cnt     = (unsigned*)take(256);

    // barrier counters must start at 0 every call (graph-capture safe)
    hipMemsetAsync(cnt, 0, 256, stream);

    // 1) embedding gather + bf16 casts + fused biases
    k_embed_gather<<<(L * E + 255) / 256, 256, 0, stream>>>(sent, embed, xs_bf);
    k_cast_bf16<<<(G * E + 255) / 256, 256, 0, stream>>>(w_ih_f, wihf_bf, G * E);
    k_cast_bf16<<<(G * E + 255) / 256, 256, 0, stream>>>(w_ih_b, wihb_bf, G * E);
    k_cast_bf16<<<(T * 2 * H + 255) / 256, 256, 0, stream>>>(w_out, wout_bf, T * 2 * H);
    k_bias_sum<<<(2 * G + 255) / 256, 256, 0, stream>>>(b_ih_f, b_hh_f, b_ih_b, b_hh_b, bias);

    // 2) x_proj GEMMs (WMMA bf16 -> f32): (4096 x 2048, K=512) per direction
    //    NT=4: 256 M-tiles x 32 N-groups = 8192 waves -> 1024 blocks of 8 waves
    k_wmma_gemm<4><<<1024, 256, 0, stream>>>(xs_bf, wihf_bf, bias, xproj,
                                             E, G, (G / 16) / 4);
    k_wmma_gemm<4><<<1024, 256, 0, stream>>>(xs_bf, wihb_bf, bias + G,
                                             xproj + (size_t)L * G,
                                             E, G, (G / 16) / 4);

    // 3) bidirectional LSTM recurrence: 16 WGs per direction, W_hh sharded into LDS
    const size_t lds = (size_t)128 * 512 * 2 + (size_t)H * 4 + 128 * 4; // ~130.5 KB
    hipFuncSetAttribute((const void*)k_lstm,
                        hipFuncAttributeMaxDynamicSharedMemorySize, (int)lds);
    k_lstm<<<32, 128, lds, stream>>>(xproj, w_hh_f, w_hh_b, h0, c0, hbuf, hs_bf, cnt);

    // 4) feats = hs @ w_out^T + b_out : (4096 x 32, K=1024)
    //    NT=2: 256 M-tiles x 1 N-group = 256 waves -> 32 blocks of 8 waves
    k_wmma_gemm<2><<<32, 256, 0, stream>>>(hs_bf, wout_bf, b_out, feats,
                                           2 * H, T, (T / 16) / 2);

    // 5) Viterbi forward + backtrace
    k_viterbi<<<1, 32, 0, stream>>>(feats, trans, bptr, (float*)d_out);
}
